// LocalAttention_56100862820998
// MI455X (gfx1250) — matrix operand order repaired and checked by reading, not compiled, not run on hardware
//
#include <hip/hip_runtime.h>
#include <hip/hip_bf16.h>
#include <cstddef>

typedef __bf16 bf16;
typedef __bf16 bf16x8 __attribute__((ext_vector_type(8)));
typedef __bf16 v16bf  __attribute__((ext_vector_type(16)));
typedef float  v8f    __attribute__((ext_vector_type(8)));

// Problem constants
constexpr int  Cdim = 1024;
constexpr int  Hn   = 16;
constexpr int  HD   = 64;
constexpr int  Tlen = 4096;
constexpr int  Bsz  = 4;
constexpr int  Mtok = Bsz * Tlen;   // 16384 token rows
constexpr int  WIN  = 64;

// ---------- WMMA fragment helpers (layouts per CDNA5 ISA §7.12.2) ----------

// A-matrix 16x32 bf16: lane l holds row M = l%16.
// elements 0..7  = K (l/16)*8 + {0..7}; elements 8..15 = K 16 + (l/16)*8 + {0..7}
// caller passes p = row_ptr + kbase + (l>>4)*8
__device__ __forceinline__ v16bf frag_a(const bf16* p) {
  bf16x8 lo = *(const bf16x8*)(p);
  bf16x8 hi = *(const bf16x8*)(p + 16);
  v16bf r;
#pragma unroll
  for (int i = 0; i < 8; ++i) { r[i] = lo[i]; r[i + 8] = hi[i]; }
  return r;
}

// B-matrix 32x16 bf16: lane l holds column N = l%16,
// elements 0..7 = K (l/16)*8 + {0..7}; elements 8..15 = K 16 + (l/16)*8 + {0..7}
// caller passes p = colsrc_row_ptr + kbase + (l>>4)*8
__device__ __forceinline__ v16bf frag_b(const bf16* p) {
  bf16x8 lo = *(const bf16x8*)(p);
  bf16x8 hi = *(const bf16x8*)(p + 16);
  v16bf r;
#pragma unroll
  for (int i = 0; i < 8; ++i) { r[i] = lo[i]; r[i + 8] = hi[i]; }
  return r;
}

__device__ __forceinline__ v8f zero8() {
  v8f z = {0.f, 0.f, 0.f, 0.f, 0.f, 0.f, 0.f, 0.f};
  return z;
}

__device__ __forceinline__ v8f wmma_bf16(v16bf a, v16bf b, v8f c) {
  return __builtin_amdgcn_wmma_f32_16x16x32_bf16(false, a, false, b, (short)0, c,
                                                 false, false);
}

// ---------------------------------------------------------------------------
// Tiled GEMM: Y[M,1024] = A[M,1024] @ W^T + bias     (W is [o][c] row-major)
// MODE 0: A = x(f32)*mask -> Q bf16 (t, o)
// MODE 1: A = x(f32)*mask -> Kraw bf16 (t, o)
// MODE 2: A = x(f32)*mask -> Vt bf16 transposed (o, t)
// MODE 3: A = AO(bf16)     -> d_out f32 (t, o)
// Block tile 64x256, BK=32, 256 threads = 8 waves in 2(M)x4(N); wave tile 32x64.
// Software pipeline: regs(k+1) global loads overlap WMMA on LDS tile(k).
// ---------------------------------------------------------------------------
template <int MODE>
__global__ __launch_bounds__(256) void gemm_kernel(
    const float* __restrict__ Af, const bf16* __restrict__ Ab,
    const unsigned char* __restrict__ pm, const float* __restrict__ Wt,
    const float* __restrict__ bias, bf16* __restrict__ outb,
    float* __restrict__ outf) {
  constexpr int LDA = 40;   // bf16 elems per LDS row (32 + pad, rows 16B-aligned)
  __shared__ bf16 As[64 * LDA];    //  5.0 KB
  __shared__ bf16 Bs[256 * LDA];   // 20.0 KB

  const int tid = threadIdx.x;
  const int m0 = blockIdx.y * 64;
  const int n0 = blockIdx.x * 256;
  const int wave = tid >> 5, lane = tid & 31;
  const int wm = wave >> 2, wn = wave & 3;
  const int ln = lane & 15, l16 = lane >> 4;

  // A-staging map: 64 rows x 32 cols, 8 elems/thread
  const int ar = tid >> 2;
  const int ac = (tid & 3) * 8;
  float pscale = 1.0f;
  if (MODE != 3) pscale = pm[m0 + ar] ? 1.0f : 0.0f;
  const float* aptr = (MODE != 3) ? Af + (size_t)(m0 + ar) * Cdim + ac : nullptr;
  const bf16* abptr = (MODE == 3) ? Ab + (size_t)(m0 + ar) * Cdim + ac : nullptr;
  // B-staging map: one W row k-slice (32 floats) per thread
  const float* wptr = Wt + (size_t)(n0 + tid) * Cdim;

  v8f acc[2][4];
#pragma unroll
  for (int tm = 0; tm < 2; ++tm)
#pragma unroll
    for (int tn = 0; tn < 4; ++tn) acc[tm][tn] = zero8();

  // ---- prologue: stage tile k0=0 into registers ----
  float4 aR0, aR1;
  bf16x8 aRb;
  float4 bR[8];
  if (MODE != 3) {
    aR0 = *(const float4*)(aptr);
    aR1 = *(const float4*)(aptr + 4);
  } else {
    aRb = *(const bf16x8*)(abptr);
  }
#pragma unroll
  for (int j = 0; j < 8; ++j) bR[j] = *(const float4*)(wptr + 4 * j);

  for (int k0 = 0; k0 < Cdim; k0 += 32) {
    // ---- commit staged registers to LDS (bf16 convert on the fly) ----
    if (MODE != 3) {
      bf16* d = As + ar * LDA + ac;
      d[0] = (bf16)(aR0.x * pscale); d[1] = (bf16)(aR0.y * pscale);
      d[2] = (bf16)(aR0.z * pscale); d[3] = (bf16)(aR0.w * pscale);
      d[4] = (bf16)(aR1.x * pscale); d[5] = (bf16)(aR1.y * pscale);
      d[6] = (bf16)(aR1.z * pscale); d[7] = (bf16)(aR1.w * pscale);
    } else {
      *(bf16x8*)(As + ar * LDA + ac) = aRb;
    }
    {
      bf16* d = Bs + tid * LDA;
#pragma unroll
      for (int j = 0; j < 8; ++j) {
        d[4 * j + 0] = (bf16)bR[j].x; d[4 * j + 1] = (bf16)bR[j].y;
        d[4 * j + 2] = (bf16)bR[j].z; d[4 * j + 3] = (bf16)bR[j].w;
      }
    }
    __syncthreads();

    // ---- issue next tile's global loads (overlaps WMMA below) ----
    const int kn = k0 + 32;
    if (kn < Cdim) {
      if (MODE != 3) {
        aR0 = *(const float4*)(aptr + kn);
        aR1 = *(const float4*)(aptr + kn + 4);
      } else {
        aRb = *(const bf16x8*)(abptr + kn);
      }
#pragma unroll
      for (int j = 0; j < 8; ++j) bR[j] = *(const float4*)(wptr + kn + 4 * j);
      // keep the streamed weight tile flowing into GL2 (global_prefetch_b8)
      if (kn + 32 < Cdim) __builtin_prefetch(wptr + kn + 32, 0, 0);
    }

    // ---- compute from LDS tile(k0): 8 wmma per wave ----
    v16bf afr[2];
#pragma unroll
    for (int tm = 0; tm < 2; ++tm)
      afr[tm] = frag_a(As + (wm * 32 + tm * 16 + ln) * LDA + l16 * 8);
#pragma unroll
    for (int tn = 0; tn < 4; ++tn) {
      const v16bf bfr = frag_b(Bs + (wn * 64 + tn * 16 + ln) * LDA + l16 * 8);
      acc[0][tn] = wmma_bf16(afr[0], bfr, acc[0][tn]);
      acc[1][tn] = wmma_bf16(afr[1], bfr, acc[1][tn]);
    }
    __syncthreads();
  }

  // ---- epilogue: C layout lane = col (l%16), VGPR v = row (l/16)*8+v ----
#pragma unroll
  for (int tm = 0; tm < 2; ++tm) {
#pragma unroll
    for (int tn = 0; tn < 4; ++tn) {
      const int gn = n0 + wn * 64 + tn * 16 + ln;
      const int gm_base = m0 + wm * 32 + tm * 16 + l16 * 8;
      const float bv = bias[gn];
      if (MODE == 2) {
        bf16x8 pack;
#pragma unroll
        for (int v = 0; v < 8; ++v) pack[v] = (bf16)(acc[tm][tn][v] + bv);
        *(bf16x8*)(outb + (size_t)gn * Mtok + gm_base) = pack;   // Vt (o, t)
      } else if (MODE == 3) {
#pragma unroll
        for (int v = 0; v < 8; ++v)
          outf[(size_t)(gm_base + v) * Cdim + gn] = acc[tm][tn][v] + bv;
      } else {
#pragma unroll
        for (int v = 0; v < 8; ++v)
          outb[(size_t)(gm_base + v) * Cdim + gn] = (bf16)(acc[tm][tn][v] + bv);
      }
    }
  }
}

// ---------------------------------------------------------------------------
// RoPE: Q in place at local pos (t%64)+32 ; Kraw -> KA (same pos) and
// KB at pos (t%64<32 ? t%64+96 : t%64-32)  [the neighbor-window position]
// one thread per (t, h, pair), pair in [0,32)
// ---------------------------------------------------------------------------
__global__ __launch_bounds__(256) void rope_kernel(bf16* __restrict__ Q,
                                                   const bf16* __restrict__ Kr,
                                                   bf16* __restrict__ KA,
                                                   bf16* __restrict__ KB) {
  const int idx = blockIdx.x * 256 + threadIdx.x;   // exact multiple
  const int pair = idx & 31;
  const int th = idx >> 5;        // t*16 + h
  const int t = th >> 4;
  const int tmod = t & (WIN - 1);
  const float invf = __expf(-(float)pair * 0.28782313662425572f); // ln(1e4)/32
  const int pq = tmod + 32;
  const int pb = (tmod < 32) ? (tmod + 96) : (tmod - 32);
  const size_t base = (size_t)th * HD + pair;

  float cq, sq, cb, sb;
  __sincosf((float)pq * invf, &sq, &cq);
  __sincosf((float)pb * invf, &sb, &cb);

  const float q1 = (float)Q[base], q2 = (float)Q[base + 32];
  Q[base]      = (bf16)(q1 * cq - q2 * sq);
  Q[base + 32] = (bf16)(q2 * cq + q1 * sq);

  const float k1 = (float)Kr[base], k2 = (float)Kr[base + 32];
  KA[base]      = (bf16)(k1 * cq - k2 * sq);
  KA[base + 32] = (bf16)(k2 * cq + k1 * sq);
  KB[base]      = (bf16)(k1 * cb - k2 * sb);
  KB[base + 32] = (bf16)(k2 * cb + k1 * sb);
}

// ---------------------------------------------------------------------------
// Attention: one wave per (window bw, head h). 64 queries x 128 keys.
// Key token for context slot s is bw*64-32+s; slots [32,96) use KA, else KB.
// S tile lives in LDS for softmax; P (bf16) refragmented as WMMA A.
// ---------------------------------------------------------------------------
__global__ __launch_bounds__(32) void attn_kernel(
    const bf16* __restrict__ Q, const bf16* __restrict__ KA,
    const bf16* __restrict__ KB, const bf16* __restrict__ Vt,
    const unsigned char* __restrict__ pm, bf16* __restrict__ AO) {
  constexpr int SST = 132;   // f32 score row stride (pad)
  constexpr int PST = 136;   // bf16 prob row stride (16B-multiple)
  __shared__ float S[64 * SST];
  __shared__ bf16 P[64 * PST];

  const int lane = threadIdx.x;
  const int ln = lane & 15, l16 = lane >> 4;
  const int h = blockIdx.x;
  const int bw = blockIdx.y;
  const int b = bw >> 6;          // nW = 64 windows per batch
  const int wloc = bw & 63;
  const int tq0 = bw * 64;        // == b*T + wloc*64

  // ---------- phase 1: S = (Q K^T) * 0.125, masked ----------
#pragma unroll
  for (int m = 0; m < 4; ++m) {
    const bf16* qrow = Q + ((size_t)(tq0 + m * 16 + ln) * Hn + h) * HD;
    const v16bf a0 = frag_a(qrow + l16 * 8);
    const v16bf a1 = frag_a(qrow + 32 + l16 * 8);
#pragma unroll
    for (int n = 0; n < 8; ++n) {
      const bf16* Kb = (n >= 2 && n < 6) ? KA : KB;
      const int key = n * 16 + ln;
      const int tl = wloc * 64 - 32 + key;              // local token index
      const int tlc = tl < 0 ? 0 : (tl > Tlen - 1 ? Tlen - 1 : tl);
      const bf16* krow = Kb + ((size_t)(b * Tlen + tlc) * Hn + h) * HD;
      v8f acc = zero8();
      acc = wmma_bf16(a0, frag_b(krow + l16 * 8), acc);
      acc = wmma_bf16(a1, frag_b(krow + 32 + l16 * 8), acc);
      const bool valid = (tl >= 0) && (tl < Tlen) && (pm[b * Tlen + tlc] != 0);
#pragma unroll
      for (int v = 0; v < 8; ++v) {
        const int q = m * 16 + l16 * 8 + v;
        S[q * SST + key] = valid ? acc[v] * 0.125f : -1.0e30f;
      }
    }
  }
  __syncthreads();

  // ---------- softmax over 128 keys, 2 rows per lane ----------
#pragma unroll
  for (int r = 0; r < 2; ++r) {
    const int q = lane + r * 32;
    float* row = S + q * SST;
    float mx = row[0];
    for (int k = 1; k < 128; ++k) mx = fmaxf(mx, row[k]);
    float sum = 0.f;
    for (int k = 0; k < 128; ++k) {
      const float e = __expf(row[k] - mx);
      row[k] = e;
      sum += e;
    }
    const float inv = 1.0f / sum;
    bf16* prow = P + q * PST;
    for (int k = 0; k < 128; ++k) prow[k] = (bf16)(row[k] * inv);
  }
  __syncthreads();

  // ---------- phase 2: O = P @ V ----------
#pragma unroll
  for (int m = 0; m < 4; ++m) {
    const bf16* prow = P + (m * 16 + ln) * PST;
    v16bf pa[4];
#pragma unroll
    for (int kc = 0; kc < 4; ++kc) pa[kc] = frag_a(prow + kc * 32 + l16 * 8);
#pragma unroll
    for (int n = 0; n < 4; ++n) {
      const int o = h * HD + n * 16 + ln;
      const bf16* vcol = Vt + (size_t)o * Mtok + (tq0 - 32);  // key t = tq0-32+s
      v8f acc = zero8();
#pragma unroll
      for (int kc = 0; kc < 4; ++kc)
        acc = wmma_bf16(pa[kc], frag_b(vcol + kc * 32 + l16 * 8), acc);
#pragma unroll
      for (int v = 0; v < 8; ++v)
        AO[(size_t)(tq0 + m * 16 + l16 * 8 + v) * Cdim + o] = (bf16)acc[v];
    }
  }
}

// ---------------------------------------------------------------------------
extern "C" void kernel_launch(void* const* d_in, const int* in_sizes, int n_in,
                              void* d_out, int out_size, void* d_ws,
                              size_t ws_size, hipStream_t stream) {
  (void)in_sizes; (void)n_in; (void)out_size; (void)ws_size;
  const float* x = (const float*)d_in[0];
  const unsigned char* pm = (const unsigned char*)d_in[1];
  const float* Wq = (const float*)d_in[2];
  const float* bq = (const float*)d_in[3];
  const float* Wk = (const float*)d_in[4];
  const float* bk = (const float*)d_in[5];
  const float* Wv = (const float*)d_in[6];
  const float* bv = (const float*)d_in[7];
  const float* Wo = (const float*)d_in[8];
  const float* bo = (const float*)d_in[9];
  float* out = (float*)d_out;

  char* ws = (char*)d_ws;
  constexpr size_t SZ = (size_t)Mtok * Cdim * sizeof(bf16);   // 32 MB
  bf16* Qb = (bf16*)(ws);            // Q (t,h,d), roped in place
  bf16* KA = (bf16*)(ws + 1 * SZ);   // K rope variant A
  bf16* KB = (bf16*)(ws + 2 * SZ);   // K rope variant B
  bf16* Vt = (bf16*)(ws + 3 * SZ);   // V transposed (h,d,t) — mid-layout so ±32-row
                                     // halo reads stay inside ws
  bf16* Kraw = (bf16*)(ws + 4 * SZ); // raw K, then reused as attention output

  const dim3 gg(Cdim / 256, Mtok / 64);   // (4, 256)
  gemm_kernel<0><<<gg, 256, 0, stream>>>(x, nullptr, pm, Wq, bq, Qb, nullptr);
  gemm_kernel<1><<<gg, 256, 0, stream>>>(x, nullptr, pm, Wk, bk, Kraw, nullptr);
  gemm_kernel<2><<<gg, 256, 0, stream>>>(x, nullptr, pm, Wv, bv, Vt, nullptr);

  rope_kernel<<<(Mtok * Hn * 32) / 256, 256, 0, stream>>>(Qb, Kraw, KA, KB);

  bf16* AO = Kraw;  // Kraw is dead after rope
  attn_kernel<<<dim3(Hn, Mtok / WIN), 32, 0, stream>>>(Qb, KA, KB, Vt, pm, AO);

  gemm_kernel<3><<<gg, 256, 0, stream>>>(nullptr, AO, nullptr, Wo, bo, nullptr, out);
}
